// DeepPowerSTGAT_11304353923667
// MI455X (gfx1250) — matrix operand order repaired and checked by reading, not compiled, not run on hardware
//
#include <hip/hip_runtime.h>
#include <hip/hip_bf16.h>
#include <math.h>

typedef _Float16 v16h __attribute__((ext_vector_type(16)));
typedef _Float16 v8h  __attribute__((ext_vector_type(8)));
typedef float    v8f  __attribute__((ext_vector_type(8)));

#define HD 64
#define G4 256
#define TSTEPS 24

__device__ __forceinline__ float sigf(float x) { return 1.0f / (1.0f + expf(-x)); }

// monotone float -> uint mapping for atomic max on floats
__device__ __forceinline__ unsigned fmap(float f) {
  unsigned u = __float_as_uint(f);
  return (u & 0x80000000u) ? ~u : (u | 0x80000000u);
}
__device__ __forceinline__ float finv(unsigned u) {
  u = (u & 0x80000000u) ? (u & 0x7fffffffu) : ~u;
  return __uint_as_float(u);
}
#define FMAP_NEG_INF 0x007FFFFFu  // fmap(-inf)

// ---- WMMA fragment loaders (16x16x32 f16 WMMA; this code's contiguous k order, same for A and B) ----
// A 16x32 f16: lane l holds row M=l&15; halves 0..15 = K kb..kb+15,
// kb = kk*32 + (l>>4)*16. Source: row-major f16 [rows][64] in LDS.
__device__ __forceinline__ v16h load_afrag(const _Float16* sA, int lane, int kk) {
  int M  = lane & 15;
  int kb = kk * 32 + (lane >> 4) * 16;
  const _Float16* p = sA + M * 64 + kb;
  v8h lo = *(const v8h*)p;
  v8h hi = *(const v8h*)(p + 8);
  v16h a;
#pragma unroll
  for (int i = 0; i < 8; ++i) { a[i] = lo[i]; a[i + 8] = hi[i]; }
  return a;
}
// B 32x16 f16: lane l holds column n = t16*16 + (l&15); halves = K kb..kb+15, kb = kk*32 + (l>>4)*16.
// Source stored transposed: sWt[n*ldk + k] = B[k][n]  -> contiguous 32B per fragment.
__device__ __forceinline__ v16h load_bfrag(const _Float16* sWt, int lane, int t16, int kk, int ldk) {
  int n  = t16 * 16 + (lane & 15);
  int kb = kk * 32 + (lane >> 4) * 16;
  return *(const v16h*)(sWt + n * ldk + kb);
}

// =============================== LSTM ===============================
// block = 256 thr = 8 waves; wave handles 16 nodes for all 24 steps.
// gates[16x256] = x_t @ Wih^T + h @ Whh^T + b, via 16 col-tiles x 2 K-steps of WMMA.
__global__ __launch_bounds__(256) void k_lstm(
    const float* __restrict__ x,    // [N,24,2]
    const float* __restrict__ Wih,  // [256,2]
    const float* __restrict__ Whh,  // [256,64]
    const float* __restrict__ bih, const float* __restrict__ bhh,
    float* __restrict__ hout, int n) {
  __shared__ _Float16 sWhh[G4 * HD] __attribute__((aligned(32)));  // sWhh[nn*64+k] = B[k][nn]
  __shared__ float sWih[G4 * 2];
  __shared__ float sB[G4];
  __shared__ _Float16 sH[8][16 * HD] __attribute__((aligned(32)));
  __shared__ float sX[8][32];

  int tid = threadIdx.x;
  for (int i = tid; i < G4 * HD; i += 256) sWhh[i] = (_Float16)Whh[i];
  for (int i = tid; i < G4 * 2; i += 256) sWih[i] = Wih[i];
  if (tid < G4) sB[tid] = bih[tid] + bhh[tid];
  for (int i = tid; i < 8 * 16 * HD; i += 256) ((_Float16*)sH)[i] = (_Float16)0.0f;
  __syncthreads();

  int wave = tid >> 5, lane = tid & 31;
  int hi = lane >> 4, lo = lane & 15;
  int rowBase = blockIdx.x * 128 + wave * 16;
  _Float16* mH = sH[wave];
  float* mX = sX[wave];

  v8f cst[4];
#pragma unroll
  for (int t = 0; t < 4; ++t) cst[t] = (v8f){};

  for (int t = 0; t < TSTEPS; ++t) {
    // stage x_t tile (16 nodes x 2) into wave-private LDS
    if (lane < 16) {
      int node = rowBase + lane;
      float x0 = 0.0f, x1 = 0.0f;
      if (node < n) { x0 = x[node * (TSTEPS * 2) + t * 2]; x1 = x[node * (TSTEPS * 2) + t * 2 + 1]; }
      mX[lane * 2] = x0; mX[lane * 2 + 1] = x1;
    }
    __builtin_amdgcn_wave_barrier();

    float xr0[8], xr1[8];
#pragma unroll
    for (int k = 0; k < 8; ++k) {
      int m = k + 8 * hi;
      xr0[k] = mX[m * 2]; xr1[k] = mX[m * 2 + 1];
    }
    v16h a0 = load_afrag(mH, lane, 0);
    v16h a1 = load_afrag(mH, lane, 1);
    __builtin_amdgcn_wave_barrier();

    v8f acc[16];
#pragma unroll
    for (int t16 = 0; t16 < 16; ++t16) {
      int nn = t16 * 16 + lo;
      float w0 = sWih[nn * 2], w1 = sWih[nn * 2 + 1], bn = sB[nn];
      v8f c;
#pragma unroll
      for (int k = 0; k < 8; ++k) c[k] = bn + w0 * xr0[k] + w1 * xr1[k];
      v16h b0 = load_bfrag(sWhh, lane, t16, 0, HD);
      c = __builtin_amdgcn_wmma_f32_16x16x32_f16(false, a0, false, b0, (short)0, c, false, false);
      v16h b1 = load_bfrag(sWhh, lane, t16, 1, HD);
      c = __builtin_amdgcn_wmma_f32_16x16x32_f16(false, a1, false, b1, (short)0, c, false, false);
      acc[t16] = c;
    }

    // pointwise gates (i,f,g,o are col-tiles t, t+4, t+8, t+12); write new h to LDS
#pragma unroll
    for (int tt = 0; tt < 4; ++tt) {
      v8f gi = acc[tt], gf = acc[tt + 4], gg = acc[tt + 8], go = acc[tt + 12];
#pragma unroll
      for (int k = 0; k < 8; ++k) {
        float c = cst[tt][k];
        c = sigf(gf[k]) * c + sigf(gi[k]) * tanhf(gg[k]);
        float hv = sigf(go[k]) * tanhf(c);
        cst[tt][k] = c;
        int m = k + 8 * hi;
        int j = tt * 16 + lo;
        mH[m * HD + j] = (_Float16)hv;
        if (t == TSTEPS - 1) {
          int node = rowBase + m;
          if (node < n) hout[node * HD + j] = hv;
        }
      }
    }
    __builtin_amdgcn_wave_barrier();
  }
}

// =============================== Generic 64x64 GEMM ===============================
// Y[row,:64] (+)= X[row,:64] @ B,  B[k][n] = W[k*skk + n*snn + off].
__global__ __launch_bounds__(256) void k_mm64(
    const float* __restrict__ X, const float* __restrict__ W, float* __restrict__ Y,
    int n, int skk, int snn, int off, int accflag) {
  __shared__ _Float16 sWt[HD * HD] __attribute__((aligned(32)));   // sWt[n*64+k] = B[k][n]
  __shared__ _Float16 sX[128 * HD] __attribute__((aligned(32)));
  int tid = threadIdx.x;
  for (int i = tid; i < HD * HD; i += 256) {
    int nn = i >> 6, k = i & 63;
    sWt[i] = (_Float16)W[k * skk + nn * snn + off];
  }
  int rowBase = blockIdx.x * 128;
  for (int i = tid; i < 128 * HD; i += 256) {
    int r = i >> 6, c = i & 63;
    int row = rowBase + r;
    sX[i] = (_Float16)(row < n ? X[row * HD + c] : 0.0f);
  }
  __syncthreads();

  int wave = tid >> 5, lane = tid & 31;
  int hi = lane >> 4, lo = lane & 15;
  const _Float16* mX = sX + wave * 16 * HD;
  int wRowBase = rowBase + wave * 16;
  v16h a0 = load_afrag(mX, lane, 0);
  v16h a1 = load_afrag(mX, lane, 1);
#pragma unroll
  for (int t = 0; t < 4; ++t) {
    v8f c = (v8f){};
    if (accflag) {
#pragma unroll
      for (int k = 0; k < 8; ++k) {
        int row = wRowBase + k + 8 * hi;
        if (row < n) c[k] = Y[row * HD + t * 16 + lo];
      }
    }
    v16h b0 = load_bfrag(sWt, lane, t, 0, HD);
    c = __builtin_amdgcn_wmma_f32_16x16x32_f16(false, a0, false, b0, (short)0, c, false, false);
    v16h b1 = load_bfrag(sWt, lane, t, 1, HD);
    c = __builtin_amdgcn_wmma_f32_16x16x32_f16(false, a1, false, b1, (short)0, c, false, false);
#pragma unroll
    for (int k = 0; k < 8; ++k) {
      int row = wRowBase + k + 8 * hi;
      if (row < n) Y[row * HD + t * 16 + lo] = c[k];
    }
  }
}

// =============================== GAT edge pipeline ===============================
__global__ void k_coef(const float* __restrict__ We, const float* __restrict__ ae, float* coef) {
  int j = threadIdx.x;
  if (j < 2) {
    float s = 0.0f;
    for (int k = 0; k < HD; ++k) s += We[j * HD + k] * ae[k];
    coef[j] = s;
  }
}

__global__ void k_scores(const float* __restrict__ hs, const float* __restrict__ as_,
                         const float* __restrict__ ad, float* __restrict__ ss,
                         float* __restrict__ sd, int n) {
  int i = blockIdx.x * blockDim.x + threadIdx.x;
  if (i >= n) return;
  float s1 = 0.0f, s2 = 0.0f;
  for (int k = 0; k < HD; ++k) {
    float v = hs[i * HD + k];
    s1 += v * as_[k];
    s2 += v * ad[k];
  }
  ss[i] = s1; sd[i] = s2;
}

__global__ void k_init(float* __restrict__ accum, unsigned* __restrict__ amax,
                       float* __restrict__ den, const float* __restrict__ bg, int n) {
  long long i = (long long)blockIdx.x * blockDim.x + threadIdx.x;
  if (i < n) { amax[i] = FMAP_NEG_INF; den[i] = 0.0f; }
  if (i < (long long)n * HD) accum[i] = bg[(int)(i & 63)];
}

__global__ void k_edge_alpha(const int* __restrict__ ei, const float* __restrict__ ea,
                             const float* __restrict__ ss, const float* __restrict__ sd,
                             const float* __restrict__ coef, float* __restrict__ alpha,
                             unsigned* __restrict__ amax, int ne) {
  int e = blockIdx.x * blockDim.x + threadIdx.x;
  if (e >= ne) return;
  int s = ei[e], d = ei[ne + e];
  float a = ss[s] + sd[d] + ea[e * 2] * coef[0] + ea[e * 2 + 1] * coef[1];
  a = a > 0.0f ? a : 0.2f * a;  // leaky relu 0.2
  alpha[e] = a;
  atomicMax(&amax[d], fmap(a));
}

__global__ void k_edge_exp(const int* __restrict__ ei, const float* __restrict__ alpha,
                           const unsigned* __restrict__ amax, float* __restrict__ ex,
                           float* __restrict__ den, int ne) {
  int e = blockIdx.x * blockDim.x + threadIdx.x;
  if (e >= ne) return;
  int d = ei[ne + e];
  float am = finv(amax[d]);
  if (!(am > -3.0e38f && am < 3.0e38f)) am = 0.0f;
  float v = expf(alpha[e] - am);
  ex[e] = v;
  atomicAdd(&den[d], v);
}

__global__ void k_edge_scatter(const int* __restrict__ ei, const float* __restrict__ ex,
                               const float* __restrict__ den, const float* __restrict__ hs,
                               float* __restrict__ accum, int ne) {
  long long idx = (long long)blockIdx.x * blockDim.x + threadIdx.x;
  if (idx >= (long long)ne * HD) return;
  int e = (int)(idx >> 6), j = (int)(idx & 63);
  int s = ei[e], d = ei[ne + e];
  float w = ex[e] / (den[d] + 1e-16f);
  atomicAdd(&accum[d * HD + j], hs[s * HD + j] * w);
}

// =============================== Graph LayerNorm ===============================
__global__ void k_ln_part(const float* __restrict__ x, float* __restrict__ part, long long tot) {
  __shared__ float s1[256], s2[256];
  float a = 0.0f, b = 0.0f;
  for (long long i = (long long)blockIdx.x * blockDim.x + threadIdx.x; i < tot;
       i += (long long)gridDim.x * blockDim.x) {
    float v = x[i];
    a += v; b += v * v;
  }
  s1[threadIdx.x] = a; s2[threadIdx.x] = b;
  __syncthreads();
  for (int o = 128; o > 0; o >>= 1) {
    if ((int)threadIdx.x < o) { s1[threadIdx.x] += s1[threadIdx.x + o]; s2[threadIdx.x] += s2[threadIdx.x + o]; }
    __syncthreads();
  }
  if (threadIdx.x == 0) { part[blockIdx.x * 2] = s1[0]; part[blockIdx.x * 2 + 1] = s2[0]; }
}

__global__ void k_ln_fin(const float* __restrict__ part, float* __restrict__ stats,
                         int nb, long long tot) {
  __shared__ float s1[256], s2[256];
  float a = 0.0f, b = 0.0f;
  for (int i = threadIdx.x; i < nb; i += 256) { a += part[i * 2]; b += part[i * 2 + 1]; }
  s1[threadIdx.x] = a; s2[threadIdx.x] = b;
  __syncthreads();
  for (int o = 128; o > 0; o >>= 1) {
    if ((int)threadIdx.x < o) { s1[threadIdx.x] += s1[threadIdx.x + o]; s2[threadIdx.x] += s2[threadIdx.x + o]; }
    __syncthreads();
  }
  if (threadIdx.x == 0) {
    float mean = s1[0] / (float)tot;
    float var  = s2[0] / (float)tot - mean * mean;
    stats[0] = mean;
    stats[1] = sqrtf(fmaxf(var, 0.0f));
  }
}

__global__ void k_ln_apply(const float* __restrict__ accum, const float* __restrict__ stats,
                           const float* __restrict__ g, const float* __restrict__ b,
                           float* __restrict__ hout, long long tot) {
  long long i = (long long)blockIdx.x * blockDim.x + threadIdx.x;
  if (i >= tot) return;
  int j = (int)(i & 63);
  float v = (accum[i] - stats[0]) / (stats[1] + 1e-5f) * g[j] + b[j];
  hout[i] = v > 0.0f ? v : 0.0f;  // fused relu
}

// =============================== Head ===============================
__global__ void k_head(const float* __restrict__ acc, const float* __restrict__ b1,
                       const float* __restrict__ W2, const float* __restrict__ b2,
                       float* __restrict__ out, int n) {
  int i = blockIdx.x * blockDim.x + threadIdx.x;
  if (i >= n) return;
  float o0 = b2[0], o1 = b2[1];
  for (int j = 0; j < HD; ++j) {
    float v = acc[i * HD + j] + b1[j];
    v = v > 0.0f ? v : 0.0f;
    o0 += v * W2[j];
    o1 += v * W2[HD + j];
  }
  out[i * 2] = o0;
  out[i * 2 + 1] = o1;
}

// =============================== launch ===============================
extern "C" void kernel_launch(void* const* d_in, const int* in_sizes, int n_in,
                              void* d_out, int out_size, void* d_ws, size_t ws_size,
                              hipStream_t stream) {
  const float* x     = (const float*)d_in[0];
  const int*   ei    = (const int*)d_in[1];
  const float* ea    = (const float*)d_in[2];
  const float* Wih   = (const float*)d_in[3];
  const float* Whh   = (const float*)d_in[4];
  const float* bih   = (const float*)d_in[5];
  const float* bhh   = (const float*)d_in[6];
  const float* Wgat  = (const float*)d_in[7];
  const float* asrc  = (const float*)d_in[8];
  const float* adst  = (const float*)d_in[9];
  const float* Wedge = (const float*)d_in[10];
  const float* aedge = (const float*)d_in[11];
  const float* bgat  = (const float*)d_in[12];
  const float* lng   = (const float*)d_in[13];
  const float* lnb   = (const float*)d_in[14];
  const float* Wfc1  = (const float*)d_in[15];
  const float* bfc1  = (const float*)d_in[16];
  const float* Wfc2  = (const float*)d_in[17];
  const float* bfc2  = (const float*)d_in[18];

  const int N = in_sizes[0] / (TSTEPS * 2);
  const int E = in_sizes[2] / 2;
  const long long NH = (long long)N * HD;

  float* ws = (float*)d_ws;
  float*    h      = ws;                 ws += NH;       // [N,64] node state
  float*    hs     = ws;                 ws += NH;       // [N,64] h @ W
  float*    accum  = ws;                 ws += NH;       // [N,64] message accumulation
  float*    accfc1 = ws;                 ws += NH;       // [N,64] JK/fc1 accumulation
  float*    ss     = ws;                 ws += N;
  float*    sd     = ws;                 ws += N;
  unsigned* amax   = (unsigned*)ws;      ws += N;
  float*    den    = ws;                 ws += N;
  float*    alpha  = ws;                 ws += E;
  float*    ex     = ws;                 ws += E;
  float*    coef   = ws;                 ws += 4;
  float*    part   = ws;                 ws += 1024;
  float*    stats  = ws;                 ws += 4;

  const int NB_LN = 512;

  k_lstm<<<(N + 127) / 128, 256, 0, stream>>>(x, Wih, Whh, bih, bhh, h, N);

  for (int l = 0; l < 5; ++l) {
    // hs = h @ W_gat[l]   (B[k][n] = W[k*64 + n])
    k_mm64<<<(N + 127) / 128, 256, 0, stream>>>(h, Wgat + (size_t)l * HD * HD, hs, N, HD, 1, 0, 0);
    k_coef<<<1, 64, 0, stream>>>(Wedge + (size_t)l * 2 * HD, aedge + (size_t)l * HD, coef);
    k_scores<<<(N + 255) / 256, 256, 0, stream>>>(hs, asrc + (size_t)l * HD, adst + (size_t)l * HD, ss, sd, N);
    k_init<<<(unsigned)((NH + 255) / 256), 256, 0, stream>>>(accum, amax, den, bgat + (size_t)l * HD, N);
    k_edge_alpha<<<(E + 255) / 256, 256, 0, stream>>>(ei, ea, ss, sd, coef, alpha, amax, E);
    k_edge_exp<<<(E + 255) / 256, 256, 0, stream>>>(ei, alpha, amax, ex, den, E);
    k_edge_scatter<<<(unsigned)(((long long)E * HD + 255) / 256), 256, 0, stream>>>(ei, ex, den, hs, accum, E);
    // graph LayerNorm + relu -> h (this is xs[l])
    k_ln_part<<<NB_LN, 256, 0, stream>>>(accum, part, NH);
    k_ln_fin<<<1, 256, 0, stream>>>(part, stats, NB_LN, NH);
    k_ln_apply<<<(unsigned)((NH + 255) / 256), 256, 0, stream>>>(accum, stats, lng + (size_t)l * HD,
                                                                 lnb + (size_t)l * HD, h, NH);
    // accfc1 (+)= xs[l] @ W_fc1[:, l*64:(l+1)*64]^T   (B[k][n] = Wfc1[n*320 + l*64 + k])
    k_mm64<<<(N + 127) / 128, 256, 0, stream>>>(h, Wfc1, accfc1, N, 1, 5 * HD, l * HD, l > 0 ? 1 : 0);
  }

  k_head<<<(N + 255) / 256, 256, 0, stream>>>(accfc1, bfc1, Wfc2, bfc2, (float*)d_out, N);
}
